// ASTHL_23742579212955
// MI455X (gfx1250) — matrix operand order repaired and checked by reading, not compiled, not run on hardware
//
#include <hip/hip_runtime.h>
#include <hip/hip_bf16.h>
#include <math.h>

typedef __attribute__((ext_vector_type(16))) _Float16 v16h;
typedef __attribute__((ext_vector_type(8)))  _Float16 v8h;
typedef __attribute__((ext_vector_type(8)))  float    v8f;

union AFrag { v16h v; v8h half8[2]; _Float16 h[16]; };
union CFrag { v8f  v; float f[8]; };

// ---------------------------------------------------------------------------
// WMMA fragment loaders from pre-converted f16 buffers (vector b128 loads)
// A 16x32 (MxK) f16: lane l holds row M=l%16; half h -> K = 16*(h>>3) + 8*(l>>4) + (h&7)
//   -> per lane: two contiguous 8-half (16 B) runs at K = kb2 and 16+kb2
// B 32x16 (KxN) f16: lane l holds col N=l%16; half h -> K = 16*(h>>3) + 8*(l>>4) + (h&7)
//   -> per lane: two contiguous 8-half (16 B) runs at K = kb and 16+kb when source is (N x K) row-major
// ---------------------------------------------------------------------------
__device__ inline v16h load_a_frag16(const _Float16* __restrict__ X, int ld, int row0, int k0, int lane) {
  AFrag a;
  int m   = lane & 15;
  int kb2 = (lane >> 4) << 3;  // 0 or 8
  const _Float16* src = X + (size_t)(row0 + m) * ld + k0 + kb2;
  a.half8[0] = *reinterpret_cast<const v8h*>(src);        // K = kb2 .. kb2+7
  a.half8[1] = *reinterpret_cast<const v8h*>(src + 16);   // K = 16+kb2 ..
  return a.v;
}

// B[k][n] = Xt[n0+n][k0+k] for Xt row-major (N x K): computes A @ Xt^T
__device__ inline v16h load_b_frag16_t(const _Float16* __restrict__ Xt, int ld, int n0, int k0, int lane) {
  AFrag b;
  int n  = lane & 15;
  int kb = (lane >> 4) << 3;  // 0 or 8
  const _Float16* src = Xt + (size_t)(n0 + n) * ld + k0 + kb;
  b.half8[0] = *reinterpret_cast<const v8h*>(src);
  b.half8[1] = *reinterpret_cast<const v8h*>(src + 16);
  return b.v;
}

// ---------------------------------------------------------------------------
// prep kernels: f32 -> f16 convert, and transpose+convert for weights
// ---------------------------------------------------------------------------
__global__ void conv_f16(const float* __restrict__ in, _Float16* __restrict__ out, int n) {
  int i = blockIdx.x * blockDim.x + threadIdx.x;
  if (i < n) out[i] = (_Float16)in[i];
}
// Wt[n*K + k] = W[k*N + n]
__global__ void transpose_f16(const float* __restrict__ W, _Float16* __restrict__ Wt, int K, int N) {
  int i = blockIdx.x * blockDim.x + threadIdx.x;
  if (i >= K * N) return;
  int k = i / N, n = i - k * N;
  Wt[(size_t)n * K + k] = (_Float16)W[i];
}

// ---------------------------------------------------------------------------
// gate: out = pe * sigmoid(pe @ W + b); 8 waves/block, one 16-row tile/wave
// ---------------------------------------------------------------------------
__global__ void __launch_bounds__(256)
gate_wmma(const _Float16* __restrict__ pe16, const float* __restrict__ pe,
          const _Float16* __restrict__ wt,   const float* __restrict__ bg,
          float* __restrict__ out, int ntiles) {
  int lane = threadIdx.x & 31;
  int tile = blockIdx.x * 8 + (threadIdx.x >> 5);
  if (tile >= ntiles) return;                       // uniform per wave
  int i0 = tile * 16;
  v16h a0 = load_a_frag16(pe16, 64, i0, 0,  lane);
  v16h a1 = load_a_frag16(pe16, 64, i0, 32, lane);
  CFrag c[4];
#pragma unroll
  for (int t = 0; t < 4; ++t)
#pragma unroll
    for (int r = 0; r < 8; ++r) c[t].f[r] = 0.f;
#pragma unroll
  for (int t = 0; t < 4; ++t) {
    v16h b0 = load_b_frag16_t(wt, 64, t * 16, 0,  lane);
    v16h b1 = load_b_frag16_t(wt, 64, t * 16, 32, lane);
    c[t].v = __builtin_amdgcn_wmma_f32_16x16x32_f16(false, a0, false, b0, (short)0, c[t].v, false, false);
    c[t].v = __builtin_amdgcn_wmma_f32_16x16x32_f16(false, a1, false, b1, (short)0, c[t].v, false, false);
  }
  int mrow = (lane >> 4) * 8;
  int ncol = lane & 15;
#pragma unroll
  for (int t = 0; t < 4; ++t)
#pragma unroll
    for (int r = 0; r < 8; ++r) {
      int row = i0 + mrow + r;
      int col = t * 16 + ncol;
      float z  = c[t].f[r] + bg[col];
      float sg = 1.0f / (1.0f + __expf(-z));
      out[(size_t)row * 64 + col] = pe[(size_t)row * 64 + col] * sg;
    }
}

// ---------------------------------------------------------------------------
// COO spmm scatter: out[rows[e]] += vals[e] * x[cols[e]]; 16 threads/nnz x 4 floats
// ---------------------------------------------------------------------------
__global__ void spmm_atomic(const int* __restrict__ rows, const int* __restrict__ cols,
                            const float* __restrict__ vals, const float* __restrict__ x,
                            float* __restrict__ out, int nnz) {
  int t = blockIdx.x * blockDim.x + threadIdx.x;
  int e = t >> 4;
  if (e >= nnz) return;
  int d = (t & 15) * 4;
  int r = rows[e], cc = cols[e];
  float v = vals[e];
  const float* xs = x + (size_t)cc * 64 + d;
  float*       os = out + (size_t)r * 64 + d;
  atomicAdd(&os[0], v * xs[0]);
  atomicAdd(&os[1], v * xs[1]);
  atomicAdd(&os[2], v * xs[2]);
  atomicAdd(&os[3], v * xs[3]);
}

// x = tmp + x ; sum += x
__global__ void prop_update(const float* __restrict__ tmp, float* __restrict__ x,
                            float* __restrict__ sum, int n) {
  int i = blockIdx.x * blockDim.x + threadIdx.x;
  if (i >= n) return;
  float xn = tmp[i] + x[i];
  x[i] = xn;
  sum[i] += xn;
}

__global__ void fill0(float* __restrict__ p, int n) {
  int i = blockIdx.x * blockDim.x + threadIdx.x;
  if (i < n) p[i] = 0.f;
}
__global__ void copyk(float* __restrict__ dst, const float* __restrict__ src, int n) {
  int i = blockIdx.x * blockDim.x + threadIdx.x;
  if (i < n) dst[i] = src[i];
}

// row-wise l2 normalize (D=64): writes f32 result AND f16 copy for WMMA use
__global__ void __launch_bounds__(32)
l2norm_rows(const float* __restrict__ in, float* __restrict__ out,
            _Float16* __restrict__ out16) {
  int row = blockIdx.x, lane = threadIdx.x;
  const float* ip = in + (size_t)row * 64;
  float a = ip[lane], b = ip[lane + 32];
  float ss = a * a + b * b;
  ss += __shfl_xor(ss, 16, 32);
  ss += __shfl_xor(ss, 8, 32);
  ss += __shfl_xor(ss, 4, 32);
  ss += __shfl_xor(ss, 2, 32);
  ss += __shfl_xor(ss, 1, 32);
  float nrm = fmaxf(sqrtf(ss), 1e-12f);
  float na = a / nrm, nb = b / nrm;
  float* op = out + (size_t)row * 64;
  op[lane] = na;
  op[lane + 32] = nb;
  _Float16* oh = out16 + (size_t)row * 64;
  oh[lane] = (_Float16)na;
  oh[lane + 32] = (_Float16)nb;
}

// ---------------------------------------------------------------------------
// InfoNCE: S = ng @ ns^T tiled 16x16 via WMMA on f16 operands; accumulate
// exp(S/T) row sums, col sums, diagonal. 8 waves/block, 16x256 strip per wave.
// ---------------------------------------------------------------------------
__global__ void __launch_bounds__(256)
infonce_tiles(const _Float16* __restrict__ ng16, const _Float16* __restrict__ ns16,
              float* __restrict__ rowsum, float* __restrict__ colsum,
              float* __restrict__ pos, int Pn) {
  int lane   = threadIdx.x & 31;
  int ntiles = (Pn + 15) >> 4;
  int itile  = blockIdx.x * 8 + (threadIdx.x >> 5);
  if (itile >= ntiles) return;                       // uniform per wave
  int i0 = itile * 16;
  int jg = blockIdx.y;
  v16h a0 = load_a_frag16(ng16, 64, i0, 0,  lane);
  v16h a1 = load_a_frag16(ng16, 64, i0, 32, lane);
  float rowAcc[8];
#pragma unroll
  for (int r = 0; r < 8; ++r) rowAcc[r] = 0.f;
  int mbase = (lane >> 4) * 8;
  int ncol  = lane & 15;
  for (int t = 0; t < 16; ++t) {
    int jt = jg * 16 + t;
    if (jt >= ntiles) break;
    int j0 = jt * 16;
    if (jt + 1 < ntiles)  // stream next B tile (global_prefetch_b8)
      __builtin_prefetch(ns16 + (size_t)(j0 + 16 + ncol) * 64, 0, 1);
    v16h b0 = load_b_frag16_t(ns16, 64, j0, 0,  lane);
    v16h b1 = load_b_frag16_t(ns16, 64, j0, 32, lane);
    CFrag c;
#pragma unroll
    for (int r = 0; r < 8; ++r) c.f[r] = 0.f;
    c.v = __builtin_amdgcn_wmma_f32_16x16x32_f16(false, a0, false, b0, (short)0, c.v, false, false);
    c.v = __builtin_amdgcn_wmma_f32_16x16x32_f16(false, a1, false, b1, (short)0, c.v, false, false);
    float colAcc = 0.f;
#pragma unroll
    for (int r = 0; r < 8; ++r) {
      float e = __expf(c.f[r] * 5.0f);  // 1/TEMP = 5
      rowAcc[r] += e;
      colAcc    += e;
      if (i0 + mbase + r == j0 + ncol) pos[i0 + mbase + r] = e;  // diagonal
    }
    colAcc += __shfl_xor(colAcc, 16, 32);
    if (lane < 16) atomicAdd(&colsum[j0 + ncol], colAcc);
  }
#pragma unroll
  for (int r = 0; r < 8; ++r) {
    float v = rowAcc[r];
    v += __shfl_xor(v, 8, 32);
    v += __shfl_xor(v, 4, 32);
    v += __shfl_xor(v, 2, 32);
    v += __shfl_xor(v, 1, 32);
    if ((lane & 15) == 0) atomicAdd(&rowsum[i0 + mbase + r], v);
  }
}

__global__ void __launch_bounds__(256)
infonce_final(const float* __restrict__ pos, const float* __restrict__ rowsum,
              const float* __restrict__ colsum, float* __restrict__ out, int Pn) {
  __shared__ float s1[256], s2[256];
  float t1 = 0.f, t2 = 0.f;
  for (int i = threadIdx.x; i < Pn; i += 256) {
    float p = pos[i];
    t1 += -__logf(p / (rowsum[i] + 1e-8f) + 1e-8f);
    t2 += -__logf(p / (colsum[i] + 1e-8f) + 1e-8f);
  }
  s1[threadIdx.x] = t1; s2[threadIdx.x] = t2;
  __syncthreads();
  for (int o = 128; o > 0; o >>= 1) {
    if (threadIdx.x < o) { s1[threadIdx.x] += s1[threadIdx.x + o]; s2[threadIdx.x] += s2[threadIdx.x + o]; }
    __syncthreads();
  }
  if (threadIdx.x == 0) out[0] = 0.5f * (s1[0] + s2[0]) / (float)Pn;
}

// msg(f16) = [g, s, p, g*s, g*p, s*p, g*s*p] per (user, dim); WMMA A-operand only
__global__ void build_msg(const float* __restrict__ mg, const float* __restrict__ ms,
                          const float* __restrict__ mp, _Float16* __restrict__ msg, int n) {
  int i = blockIdx.x * blockDim.x + threadIdx.x;
  if (i >= n) return;
  int u = i >> 6, d = i & 63;
  float g = mg[i], s = ms[i], p = mp[i];
  _Float16* row = msg + (size_t)u * 448;
  row[d]       = (_Float16)g;
  row[64 + d]  = (_Float16)s;
  row[128 + d] = (_Float16)p;
  row[192 + d] = (_Float16)(g * s);
  row[256 + d] = (_Float16)(g * p);
  row[320 + d] = (_Float16)(s * p);
  row[384 + d] = (_Float16)(g * s * p);
}

// msg_emb = msg @ fusion_w + fusion_b ; hgu = me + ue + me*ue (fused epilogue)
// fwt is fusion_w transposed to (64 x 448) f16
__global__ void __launch_bounds__(256)
fusion_wmma(const _Float16* __restrict__ msg, const _Float16* __restrict__ fwt,
            const float* __restrict__ fb, const float* __restrict__ ue,
            float* __restrict__ hgu, int ntiles) {
  int lane = threadIdx.x & 31;
  int tile = blockIdx.x * 8 + (threadIdx.x >> 5);
  if (tile >= ntiles) return;                       // uniform per wave
  int i0 = tile * 16;
  CFrag c[4];
#pragma unroll
  for (int t = 0; t < 4; ++t)
#pragma unroll
    for (int r = 0; r < 8; ++r) c[t].f[r] = 0.f;
  for (int kt = 0; kt < 14; ++kt) {
    v16h a = load_a_frag16(msg, 448, i0, kt * 32, lane);
#pragma unroll
    for (int t = 0; t < 4; ++t) {
      v16h b = load_b_frag16_t(fwt, 448, t * 16, kt * 32, lane);
      c[t].v = __builtin_amdgcn_wmma_f32_16x16x32_f16(false, a, false, b, (short)0, c[t].v, false, false);
    }
  }
  int mrow = (lane >> 4) * 8;
  int ncol = lane & 15;
#pragma unroll
  for (int t = 0; t < 4; ++t)
#pragma unroll
    for (int r = 0; r < 8; ++r) {
      int row = i0 + mrow + r;
      int col = t * 16 + ncol;
      float me = c[t].f[r] + fb[col];
      float u  = ue[(size_t)row * 64 + col];
      hgu[(size_t)row * 64 + col] = me + u + me * u;
    }
}

// fusion_pois = l2norm(hg_pois) + ng + ns (one wave per row)
__global__ void __launch_bounds__(32)
fuse_pois(const float* __restrict__ hgp, const float* __restrict__ ng,
          const float* __restrict__ ns, float* __restrict__ out) {
  int row = blockIdx.x, lane = threadIdx.x;
  const float* ip = hgp + (size_t)row * 64;
  float a = ip[lane], b = ip[lane + 32];
  float ss = a * a + b * b;
  ss += __shfl_xor(ss, 16, 32);
  ss += __shfl_xor(ss, 8, 32);
  ss += __shfl_xor(ss, 4, 32);
  ss += __shfl_xor(ss, 2, 32);
  ss += __shfl_xor(ss, 1, 32);
  float nrm = fmaxf(sqrtf(ss), 1e-12f);
  size_t o = (size_t)row * 64;
  out[o + lane]      = a / nrm + ng[o + lane]      + ns[o + lane];
  out[o + lane + 32] = b / nrm + ng[o + lane + 32] + ns[o + lane + 32];
}

// batch_users = l2norm(users_struct[user_idx]) (one wave per batch row)
__global__ void __launch_bounds__(32)
batch_users_k(const float* __restrict__ us, const int* __restrict__ idx,
              float* __restrict__ out) {
  int b = blockIdx.x, lane = threadIdx.x;
  int row = idx[b];
  const float* ip = us + (size_t)row * 64;
  float a = ip[lane], c = ip[lane + 32];
  float ss = a * a + c * c;
  ss += __shfl_xor(ss, 16, 32);
  ss += __shfl_xor(ss, 8, 32);
  ss += __shfl_xor(ss, 4, 32);
  ss += __shfl_xor(ss, 2, 32);
  ss += __shfl_xor(ss, 1, 32);
  float nrm = fmaxf(sqrtf(ss), 1e-12f);
  float* op = out + (size_t)b * 64;
  op[lane] = a / nrm;
  op[lane + 32] = c / nrm;
}

// ---------------------------------------------------------------------------
extern "C" void kernel_launch(void* const* d_in, const int* in_sizes, int n_in,
                              void* d_out, int out_size, void* d_ws, size_t ws_size,
                              hipStream_t stream) {
  const float* poi   = (const float*)d_in[0];
  const float* uemb  = (const float*)d_in[1];
  const float* w_geo = (const float*)d_in[2];
  const float* b_geo = (const float*)d_in[3];
  const float* w_seq = (const float*)d_in[4];
  const float* b_seq = (const float*)d_in[5];
  const float* w_col = (const float*)d_in[6];
  const float* b_col = (const float*)d_in[7];
  const float* fw    = (const float*)d_in[8];
  const float* fb    = (const float*)d_in[9];
  const float* geo_v = (const float*)d_in[10];
  const float* src_v = (const float*)d_in[11];
  const float* tar_v = (const float*)d_in[12];
  const float* up_v  = (const float*)d_in[13];
  const float* pu_v  = (const float*)d_in[14];
  const int* geo_r = (const int*)d_in[15];
  const int* geo_c = (const int*)d_in[16];
  const int* src_r = (const int*)d_in[17];
  const int* src_c = (const int*)d_in[18];
  const int* tar_r = (const int*)d_in[19];
  const int* tar_c = (const int*)d_in[20];
  const int* up_r  = (const int*)d_in[21];
  const int* up_c  = (const int*)d_in[22];
  const int* pu_r  = (const int*)d_in[23];
  const int* pu_c  = (const int*)d_in[24];
  const int* uidx  = (const int*)d_in[25];

  const int P  = in_sizes[0] / 64 - 1;    // 10000
  const int U  = in_sizes[1] / 64;        // 8000
  const int Bn = in_sizes[25];            // 4096
  const int NG  = in_sizes[10], ND = in_sizes[11];
  const int NU  = in_sizes[13], NPU = in_sizes[14];
  const int PD  = P * 64, UD = U * 64;
  const int Pt  = P / 16, Ut = U / 16;    // 625, 500 (both exact)

  float* out      = (float*)d_out;
  float* bu_out   = out;
  float* fp_out   = out + (size_t)Bn * 64;
  float* loss_out = fp_out + (size_t)P * 64;

  // ---- workspace carve: f32 region, then f16 region (16B-aligned sizes) ---
  float* w = (float*)d_ws;
  float* geo_g = w;  w += PD;
  float* seq_g = w;  w += PD;
  float* col_g = w;  w += PD;
  float* xbuf  = w;  w += PD;
  float* tmp   = w;  w += PD;
  float* sum   = w;  w += PD;
  float* tmp2  = w;  w += PD;
  float* ng    = w;  w += PD;
  float* ns    = w;  w += PD;
  float* hgp   = w;  w += PD;
  float* mgeo  = w;  w += UD;
  float* mseq  = w;  w += UD;
  float* mpoi  = w;  w += UD;
  float* hgu   = w;  w += UD;
  float* usb   = w;  w += UD;
  float* rowsum = w; w += P;
  float* colsum = w; w += P;
  float* pos    = w; w += ((P + 3) & ~3);   // keep following f16 region aligned
  _Float16* h = (_Float16*)w;
  _Float16* pe16  = h;  h += PD;
  _Float16* ng16  = h;  h += PD;
  _Float16* ns16  = h;  h += PD;
  _Float16* wt_g  = h;  h += 64 * 64;
  _Float16* wt_s  = h;  h += 64 * 64;
  _Float16* wt_c  = h;  h += 64 * 64;
  _Float16* fwt   = h;  h += 448 * 64;
  _Float16* msg16 = h;  h += (size_t)U * 448;

  const int EB = 256;
  auto gs = [](int n, int b) { return (n + b - 1) / b; };

  // ---- prep: f16 operands for every WMMA consumer -------------------------
  conv_f16<<<gs(PD, EB), EB, 0, stream>>>(poi, pe16, PD);
  transpose_f16<<<gs(64 * 64, EB), EB, 0, stream>>>(w_geo, wt_g, 64, 64);
  transpose_f16<<<gs(64 * 64, EB), EB, 0, stream>>>(w_seq, wt_s, 64, 64);
  transpose_f16<<<gs(64 * 64, EB), EB, 0, stream>>>(w_col, wt_c, 64, 64);
  transpose_f16<<<gs(448 * 64, EB), EB, 0, stream>>>(fw, fwt, 448, 64);

  // ---- gates: g = pe * sigmoid(pe @ W + b) --------------------------------
  gate_wmma<<<gs(Pt, 8), 256, 0, stream>>>(pe16, poi, wt_g, b_geo, geo_g, Pt);
  gate_wmma<<<gs(Pt, 8), 256, 0, stream>>>(pe16, poi, wt_s, b_seq, seq_g, Pt);
  gate_wmma<<<gs(Pt, 8), 256, 0, stream>>>(pe16, poi, wt_c, b_col, col_g, Pt);

  // ---- geo propagation: x_{l+1} = G x_l + x_l -----------------------------
  copyk<<<gs(PD, EB), EB, 0, stream>>>(xbuf, geo_g, PD);
  copyk<<<gs(PD, EB), EB, 0, stream>>>(sum,  geo_g, PD);
  for (int l = 0; l < 2; ++l) {
    fill0<<<gs(PD, EB), EB, 0, stream>>>(tmp, PD);
    spmm_atomic<<<gs(NG * 16, EB), EB, 0, stream>>>(geo_r, geo_c, geo_v, xbuf, tmp, NG);
    prop_update<<<gs(PD, EB), EB, 0, stream>>>(tmp, xbuf, sum, PD);
  }
  l2norm_rows<<<P, 32, 0, stream>>>(sum, ng, ng16);  // l2norm(sum/3)==l2norm(sum)

  // ---- seq propagation: x_{l+1} = S (T x_l) + x_l -------------------------
  copyk<<<gs(PD, EB), EB, 0, stream>>>(xbuf, seq_g, PD);
  copyk<<<gs(PD, EB), EB, 0, stream>>>(sum,  seq_g, PD);
  for (int l = 0; l < 2; ++l) {
    fill0<<<gs(PD, EB), EB, 0, stream>>>(tmp2, PD);
    spmm_atomic<<<gs(ND * 16, EB), EB, 0, stream>>>(tar_r, tar_c, tar_v, xbuf, tmp2, ND);
    fill0<<<gs(PD, EB), EB, 0, stream>>>(tmp, PD);
    spmm_atomic<<<gs(ND * 16, EB), EB, 0, stream>>>(src_r, src_c, src_v, tmp2, tmp, ND);
    prop_update<<<gs(PD, EB), EB, 0, stream>>>(tmp, xbuf, sum, PD);
  }
  l2norm_rows<<<P, 32, 0, stream>>>(sum, ns, ns16);

  // ---- InfoNCE (both directions from one tiled pass over S = ng @ ns^T) ---
  fill0<<<gs(P, EB), EB, 0, stream>>>(rowsum, P);
  fill0<<<gs(P, EB), EB, 0, stream>>>(colsum, P);
  infonce_tiles<<<dim3(gs(Pt, 8), (Pt + 15) / 16), 256, 0, stream>>>(ng16, ns16, rowsum, colsum, pos, P);
  infonce_final<<<1, 256, 0, stream>>>(pos, rowsum, colsum, loss_out, P);

  // ---- MV layers (loop-invariant: hg_pois = col_g + spmm(pu, hg_users)) ---
  fill0<<<gs(UD, EB), EB, 0, stream>>>(mgeo, UD);
  fill0<<<gs(UD, EB), EB, 0, stream>>>(mseq, UD);
  fill0<<<gs(UD, EB), EB, 0, stream>>>(mpoi, UD);
  spmm_atomic<<<gs(NU * 16, EB), EB, 0, stream>>>(up_r, up_c, up_v, ng,    mgeo, NU);
  spmm_atomic<<<gs(NU * 16, EB), EB, 0, stream>>>(up_r, up_c, up_v, ns,    mseq, NU);
  spmm_atomic<<<gs(NU * 16, EB), EB, 0, stream>>>(up_r, up_c, up_v, col_g, mpoi, NU);
  build_msg<<<gs(UD, EB), EB, 0, stream>>>(mgeo, mseq, mpoi, msg16, UD);
  fusion_wmma<<<gs(Ut, 8), 256, 0, stream>>>(msg16, fwt, fb, uemb, hgu, Ut);
  copyk<<<gs(PD, EB), EB, 0, stream>>>(hgp, col_g, PD);
  spmm_atomic<<<gs(NPU * 16, EB), EB, 0, stream>>>(pu_r, pu_c, pu_v, hgu, hgp, NPU);

  // ---- outputs ------------------------------------------------------------
  fuse_pois<<<P, 32, 0, stream>>>(hgp, ng, ns, fp_out);
  fill0<<<gs(UD, EB), EB, 0, stream>>>(usb, UD);
  spmm_atomic<<<gs(NU * 16, EB), EB, 0, stream>>>(up_r, up_c, up_v, fp_out, usb, NU);
  batch_users_k<<<Bn, 32, 0, stream>>>(usb, uidx, bu_out);
}
